// OmegaWLPWavLM_42502996361281
// MI455X (gfx1250) — compile-verified, matrix operand order checked by reading, not run on hardware
//
#include <hip/hip_runtime.h>
#include <hip/hip_bf16.h>
#include <math.h>

// ---------------- types for WMMA ----------------
typedef __attribute__((ext_vector_type(16))) __bf16 v16bf;
typedef __attribute__((ext_vector_type(8)))  float  v8f;

__device__ __forceinline__ __bf16 f2bf(float f) {
  unsigned u = __builtin_bit_cast(unsigned, f);
  unsigned r = u + 0x7FFFu + ((u >> 16) & 1u);           // round-to-nearest-even
  unsigned short h = (unsigned short)(r >> 16);
  return __builtin_bit_cast(__bf16, h);
}
__device__ __forceinline__ unsigned short bfbits(float f) {
  unsigned u = __builtin_bit_cast(unsigned, f);
  return (unsigned short)((u + 0x7FFFu + ((u >> 16) & 1u)) >> 16);
}

__device__ __forceinline__ float gelu_f(float x) {
  return 0.5f * x * (1.0f + erff(x * 0.70710678118654752f));
}
__device__ __forceinline__ float sigmoid_f(float x) { return 1.0f / (1.0f + expf(-x)); }

// ---------------- model constants ----------------
#define BN 8
#define LW 80000
#define DM 768
#define HN 12
#define HD 64
#define FFN 3072
#define NL 12
#define CD 512
#define TT 249
#define ROWS (BN * TT)          // 1992
#define NBUCK 320

// ---------------- WMMA fragment helpers ----------------
// lane layout (ISA 7.12.2, 16-bit A 16x32): lanes<16 -> K {0..7,16..23},
// lanes>=16 -> K {8..15,24..31}; kbase = (lane>=16)*8
__device__ __forceinline__ v16bf frag_bt(const float* __restrict__ B, int ldb,
                                         int bn, int N, int K, int k0, int kbase) {
  v16bf f;
#pragma unroll
  for (int e = 0; e < 16; ++e) {
    int kk = k0 + kbase + (e < 8 ? e : e + 8);
    float v = (bn < N && kk < K) ? B[(size_t)bn * ldb + kk] : 0.f;
    f[e] = f2bf(v);
  }
  return f;
}
// fragment from an LDS tile stored as [row][32 k] bf16, 64B rows:
// two 16B (ds_load_b128) reads per fragment
__device__ __forceinline__ v16bf lds_frag(const unsigned short (*S)[32], int row, int kbase) {
  union { v16bf v; uint4 q[2]; } u;
  const uint4* p = (const uint4*)S[row];
  u.q[0] = p[kbase >> 3];
  u.q[1] = p[(kbase >> 3) + 2];
  return u.v;
}
// store one 16x16 accumulator tile with epilogue
__device__ __forceinline__ void store_tile(float* __restrict__ C, int ldc,
                                           const float* __restrict__ bias,
                                           v8f acc, int mb, int nb, int half, int mr,
                                           int M, int N, int act, float alpha) {
#pragma unroll
  for (int i = 0; i < 8; ++i) {
    int m = mb + half * 8 + i;
    int n = nb + mr;
    if (m < M && n < N) {
      float r = acc[i];
      if (bias) r += bias[n];
      r *= alpha;
      if (act == 1) r = gelu_f(r);
      else if (act == 2) r = fmaxf(r, 0.f);
      C[(size_t)m * ldc + n] = r;
    }
  }
}

#define WMMA_BF16(A_, B_, C_) \
  __builtin_amdgcn_wmma_f32_16x16x32_bf16(false, (A_), false, (B_), (short)0, (C_), false, false)

// =====================================================================
// 1) waveform normalization: per batch, mean / std(ddof=1)
// =====================================================================
__global__ void k_wav_norm(const float* __restrict__ wav, float* __restrict__ out, int n) {
  int b = blockIdx.x, tid = threadIdx.x;
  const float* w = wav + (size_t)b * n;
  float* o = out + (size_t)b * n;
  __shared__ float red[256];
  float s = 0.f;
  for (int i = tid; i < n; i += 256) s += w[i];
  red[tid] = s; __syncthreads();
  for (int st = 128; st > 0; st >>= 1) { if (tid < st) red[tid] += red[tid + st]; __syncthreads(); }
  float mean = red[0] / n; __syncthreads();
  float v = 0.f;
  for (int i = tid; i < n; i += 256) { float d = w[i] - mean; v += d * d; }
  red[tid] = v; __syncthreads();
  for (int st = 128; st > 0; st >>= 1) { if (tid < st) red[tid] += red[tid + st]; __syncthreads(); }
  float inv = 1.0f / (sqrtf(red[0] / (float)(n - 1)) + 1e-6f);
  for (int i = tid; i < n; i += 256) o[i] = (w[i] - mean) * inv;
}

// =====================================================================
// 2) conv0: k=10, s=5, 1->512 channels (tiny K, direct)
// =====================================================================
__global__ void k_conv0(const float* __restrict__ x, const float* __restrict__ w,
                        float* __restrict__ y, int tin, int tout) {
  int t = blockIdx.x * blockDim.x + threadIdx.x;
  int c = blockIdx.y, b = blockIdx.z;
  if (t >= tout) return;
  const float* xi = x + (size_t)b * tin + (size_t)t * 5;
  const float* wc = w + (size_t)c * 10;
  float s = 0.f;
#pragma unroll
  for (int j = 0; j < 10; ++j) s += wc[j] * xi[j];
  y[((size_t)b * CD + c) * tout + t] = s;
}

// =====================================================================
// 3) instance norm over time (per b,c) + gelu, in place
// =====================================================================
__global__ void k_instnorm_gelu(float* __restrict__ x, const float* __restrict__ gw,
                                const float* __restrict__ gb, int n) {
  int bc = blockIdx.x, tid = threadIdx.x;
  int c = bc & (CD - 1);
  float* p = x + (size_t)bc * n;
  __shared__ float red[256];
  float s = 0.f;
  for (int i = tid; i < n; i += 256) s += p[i];
  red[tid] = s; __syncthreads();
  for (int st = 128; st > 0; st >>= 1) { if (tid < st) red[tid] += red[tid + st]; __syncthreads(); }
  float mean = red[0] / n; __syncthreads();
  float v = 0.f;
  for (int i = tid; i < n; i += 256) { float d = p[i] - mean; v += d * d; }
  red[tid] = v; __syncthreads();
  for (int st = 128; st > 0; st >>= 1) { if (tid < st) red[tid] += red[tid + st]; __syncthreads(); }
  float sc = rsqrtf(red[0] / n + 1e-5f) * gw[c];
  float sh = gb[c];
  for (int i = tid; i < n; i += 256) p[i] = gelu_f((p[i] - mean) * sc + sh);
}

// =====================================================================
// 4) conv layers 2..7 as im2col-on-the-fly WMMA GEMM + gelu
//    one wave computes a 32x32 (t, out_ch) tile: 2x2 WMMA accumulators
// =====================================================================
__global__ void k_conv_wmma_gelu(const float* __restrict__ x, const float* __restrict__ w,
                                 float* __restrict__ y, int tin, int tout, int kw, int stride) {
  int lane = threadIdx.x;
  int m0 = blockIdx.x * 32, n0 = blockIdx.y * 32, b = blockIdx.z;
  int K = CD * kw;
  const float* xb = x + (size_t)b * CD * tin;
  int half = lane >> 4, mr = lane & 15;
  int kbase = half * 8;
  int am0 = m0 + mr, am1 = m0 + 16 + mr;       // output time rows this lane loads
  int bn0 = n0 + mr, bn1 = n0 + 16 + mr;       // output channels this lane loads
  v8f c00 = {}, c01 = {}, c10 = {}, c11 = {};
#pragma unroll 2
  for (int k0 = 0; k0 < K; k0 += 32) {
    v16bf a0, a1;
#pragma unroll
    for (int e = 0; e < 16; ++e) {
      int kk = k0 + kbase + (e < 8 ? e : e + 8);
      float v0 = 0.f, v1 = 0.f;
      if (kk < K) {
        int ci = kk / kw, j = kk - ci * kw;
        const float* base = xb + (size_t)ci * tin + j;
        if (am0 < tout) v0 = base[(size_t)am0 * stride];
        if (am1 < tout) v1 = base[(size_t)am1 * stride];
      }
      a0[e] = f2bf(v0);
      a1[e] = f2bf(v1);
    }
    v16bf b0 = frag_bt(w, K, bn0, CD, K, k0, kbase);   // weights stored [outC, K]
    v16bf b1 = frag_bt(w, K, bn1, CD, K, k0, kbase);
    c00 = WMMA_BF16(a0, b0, c00);
    c01 = WMMA_BF16(a0, b1, c01);
    c10 = WMMA_BF16(a1, b0, c10);
    c11 = WMMA_BF16(a1, b1, c11);
  }
  // output layout y[b, n, m]  (NCH) with gelu
#pragma unroll
  for (int i = 0; i < 8; ++i) {
    int mA = m0 + half * 8 + i;
    int mB = mA + 16;
    int nA = n0 + mr;
    int nB = nA + 16;
    if (mA < tout) {
      y[((size_t)b * CD + nA) * tout + mA] = gelu_f(c00[i]);
      y[((size_t)b * CD + nB) * tout + mA] = gelu_f(c01[i]);
    }
    if (mB < tout) {
      y[((size_t)b * CD + nA) * tout + mB] = gelu_f(c10[i]);
      y[((size_t)b * CD + nB) * tout + mB] = gelu_f(c11[i]);
    }
  }
}

// =====================================================================
// 5) transpose conv output [b, 512, T] -> rows [b*T+t, 512]
// =====================================================================
__global__ void k_transpose_ct(const float* __restrict__ x, float* __restrict__ y, int T) {
  int c = threadIdx.x;               // 512
  int t = blockIdx.x, b = blockIdx.y;
  y[((size_t)b * T + t) * CD + c] = x[((size_t)b * CD + c) * T + t];
}

// =====================================================================
// 6) LayerNorm over last dim (optional fused residual, optional relu)
// =====================================================================
__global__ void k_layernorm(const float* __restrict__ x, const float* __restrict__ res,
                            const float* __restrict__ w, const float* __restrict__ b,
                            float* __restrict__ out, int D, int act) {
  int row = blockIdx.x, tid = threadIdx.x;
  const float* xr = x + (size_t)row * D;
  const float* rr = res ? res + (size_t)row * D : nullptr;
  float* o = out + (size_t)row * D;
  __shared__ float red[256];
  float s = 0.f;
  for (int i = tid; i < D; i += 256) { float v = xr[i]; if (rr) v += rr[i]; s += v; }
  red[tid] = s; __syncthreads();
  for (int st = 128; st > 0; st >>= 1) { if (tid < st) red[tid] += red[tid + st]; __syncthreads(); }
  float mean = red[0] / D; __syncthreads();
  float vs = 0.f;
  for (int i = tid; i < D; i += 256) {
    float v = xr[i]; if (rr) v += rr[i];
    v -= mean; vs += v * v;
  }
  red[tid] = vs; __syncthreads();
  for (int st = 128; st > 0; st >>= 1) { if (tid < st) red[tid] += red[tid + st]; __syncthreads(); }
  float inv = rsqrtf(red[0] / D + 1e-5f);
  for (int i = tid; i < D; i += 256) {
    float v = xr[i]; if (rr) v += rr[i];
    v = (v - mean) * inv * w[i] + b[i];
    if (act == 2) v = fmaxf(v, 0.f);
    o[i] = v;
  }
}

// =====================================================================
// 7) LDS-tiled, double-buffered, batched bf16-WMMA GEMM.
//    Block = 128 threads (4 waves) computing a 64x64 C tile.
//    Ping-pong LDS tiles: staging of K-slice k+32 overlaps WMMA on slice k,
//    one barrier per 32-K step. Fragments read via two ds_load_b128 each.
//    C = act((A x B + bias) * alpha); per-z offset = zo*s?o + zi*s?i
// =====================================================================
__global__ void k_gemm(const float* __restrict__ Am, int lda, long long sAi, long long sAo,
                       const float* __restrict__ Bm, int ldb, long long sBi, long long sBo,
                       int transB, const float* __restrict__ bias,
                       float* __restrict__ Cm, int ldc, long long sCi, long long sCo,
                       int M, int N, int K, int inner, int act, float alpha) {
  __shared__ __align__(16) unsigned short As[2][64][32];   // [buf][m_local][k_local]
  __shared__ __align__(16) unsigned short Bs[2][64][32];   // [buf][n_local][k_local]
  int tid = threadIdx.x;
  int lane = tid & 31;
  int w = tid >> 5;                 // wave 0..3
  int wm = w >> 1, wn = w & 1;      // 2x2 wave grid -> 32x32 each
  int z = blockIdx.z;
  int zi = z % inner, zo = z / inner;
  Am += zo * sAo + zi * sAi;
  Bm += zo * sBo + zi * sBi;
  Cm += zo * sCo + zi * sCi;
  int m0 = blockIdx.x * 64, n0 = blockIdx.y * 64;
  int half = lane >> 4, mr = lane & 15;
  int kbase = half * 8;

  // staging assignments
  int rA = tid >> 1;                 // 0..63 : A row / B(transB) row
  int hA = (tid & 1) * 16;           // 0/16  : k half
  int kB = tid >> 2;                 // 0..31 : B(k-major) k row
  int nB = (tid & 3) * 16;           // 0..48 : n quarter

  // stage one 32-wide K slice into LDS buffer `buf`
  auto stage = [&](int buf, int k0) {
    {  // A tile
      int gm = m0 + rA;
      bool fast = (gm < M) && (k0 + hA + 16 <= K) && ((lda & 3) == 0);
      if (fast) {
        const float4* src = (const float4*)&Am[(size_t)gm * lda + k0 + hA];
#pragma unroll
        for (int q = 0; q < 4; ++q) {
          float4 f = src[q];
          As[buf][rA][hA + q * 4 + 0] = bfbits(f.x);
          As[buf][rA][hA + q * 4 + 1] = bfbits(f.y);
          As[buf][rA][hA + q * 4 + 2] = bfbits(f.z);
          As[buf][rA][hA + q * 4 + 3] = bfbits(f.w);
        }
      } else {
#pragma unroll
        for (int e = 0; e < 16; ++e) {
          int kk = k0 + hA + e;
          float v = (gm < M && kk < K) ? Am[(size_t)gm * lda + kk] : 0.f;
          As[buf][rA][hA + e] = bfbits(v);
        }
      }
    }
    if (transB) {  // B stored NxK: same pattern as A
      int gn = n0 + rA;
      bool fast = (gn < N) && (k0 + hA + 16 <= K) && ((ldb & 3) == 0);
      if (fast) {
        const float4* src = (const float4*)&Bm[(size_t)gn * ldb + k0 + hA];
#pragma unroll
        for (int q = 0; q < 4; ++q) {
          float4 f = src[q];
          Bs[buf][rA][hA + q * 4 + 0] = bfbits(f.x);
          Bs[buf][rA][hA + q * 4 + 1] = bfbits(f.y);
          Bs[buf][rA][hA + q * 4 + 2] = bfbits(f.z);
          Bs[buf][rA][hA + q * 4 + 3] = bfbits(f.w);
        }
      } else {
#pragma unroll
        for (int e = 0; e < 16; ++e) {
          int kk = k0 + hA + e;
          float v = (gn < N && kk < K) ? Bm[(size_t)gn * ldb + kk] : 0.f;
          Bs[buf][rA][hA + e] = bfbits(v);
        }
      }
    } else {       // B stored KxN: coalesced along n, LDS transposed to [n][k]
      int kkg = k0 + kB;
      bool fast = (kkg < K) && ((ldb & 3) == 0) && (n0 + nB + 16 <= N);
      if (fast) {
        const float4* src = (const float4*)&Bm[(size_t)kkg * ldb + n0 + nB];
#pragma unroll
        for (int q = 0; q < 4; ++q) {
          float4 f = src[q];
          Bs[buf][nB + q * 4 + 0][kB] = bfbits(f.x);
          Bs[buf][nB + q * 4 + 1][kB] = bfbits(f.y);
          Bs[buf][nB + q * 4 + 2][kB] = bfbits(f.z);
          Bs[buf][nB + q * 4 + 3][kB] = bfbits(f.w);
        }
      } else {
#pragma unroll
        for (int e = 0; e < 16; ++e) {
          int n = n0 + nB + e;
          float v = (kkg < K && n < N) ? Bm[(size_t)kkg * ldb + n] : 0.f;
          Bs[buf][nB + e][kB] = bfbits(v);
        }
      }
    }
  };

  v8f c00 = {}, c01 = {}, c10 = {}, c11 = {};
  // 2x2 WMMA over one staged buffer
  auto compute = [&](int buf) {
    v16bf a0 = lds_frag(As[buf], wm * 32 + mr, kbase);
    v16bf a1 = lds_frag(As[buf], wm * 32 + 16 + mr, kbase);
    v16bf b0 = lds_frag(Bs[buf], wn * 32 + mr, kbase);
    v16bf b1 = lds_frag(Bs[buf], wn * 32 + 16 + mr, kbase);
    c00 = WMMA_BF16(a0, b0, c00);
    c01 = WMMA_BF16(a0, b1, c01);
    c10 = WMMA_BF16(a1, b0, c10);
    c11 = WMMA_BF16(a1, b1, c11);
  };

  // software pipeline: stage(0) ; { stage(1) | compute(0) } ; { stage(0) | compute(1) } ...
  stage(0, 0);
  __syncthreads();
  for (int k0 = 0; k0 < K; k0 += 64) {
    if (k0 + 32 < K) stage(1, k0 + 32);
    compute(0);
    __syncthreads();
    if (k0 + 32 < K) {
      if (k0 + 64 < K) stage(0, k0 + 64);
      compute(1);
      __syncthreads();
    }
  }

  int mw = m0 + wm * 32, nw = n0 + wn * 32;
  store_tile(Cm, ldc, bias, c00, mw,      nw,      half, mr, M, N, act, alpha);
  store_tile(Cm, ldc, bias, c01, mw,      nw + 16, half, mr, M, N, act, alpha);
  store_tile(Cm, ldc, bias, c10, mw + 16, nw,      half, mr, M, N, act, alpha);
  store_tile(Cm, ldc, bias, c11, mw + 16, nw + 16, half, mr, M, N, act, alpha);
}

// =====================================================================
// 8) grouped positional conv (768 ch, 16 groups, k=128, pad 64) + gelu
// =====================================================================
__global__ void k_posconv_gelu(const float* __restrict__ h, const float* __restrict__ w,
                               const float* __restrict__ bias, float* __restrict__ pc, int T) {
  int c = threadIdx.x;                     // 0..767
  int t = blockIdx.x, b = blockIdx.y;
  int g = c / 48;
  float s = bias[c];
  const float* wc = w + (size_t)c * 48 * 128;
  for (int ci = 0; ci < 48; ++ci) {
    int cin = g * 48 + ci;
    const float* wr = wc + (size_t)ci * 128;
    for (int j = 0; j < 128; ++j) {
      int tt = t + j - 64;
      if (tt >= 0 && tt < T) s += wr[j] * h[((size_t)b * T + tt) * DM + cin];
    }
  }
  pc[((size_t)b * T + t) * DM + c] = gelu_f(s);
}

// =====================================================================
// 9) relative position bias buckets -> pb[h, i, j]
// =====================================================================
__global__ void k_posbias(const float* __restrict__ re, float* __restrict__ pb, int T) {
  int j = blockIdx.x * blockDim.x + threadIdx.x;
  int i = blockIdx.y;
  if (j >= T) return;
  int rel = j - i;
  const int nb = NBUCK / 2;        // 160
  const int me = nb / 2;           // 80
  int bucket = rel > 0 ? nb : 0;
  int ar = rel < 0 ? -rel : rel;
  int val;
  if (ar < me) val = ar;
  else {
    float l = logf((float)ar / (float)me) / logf(800.0f / (float)me) * (float)(nb - me);
    int large = me + (int)l;
    val = large < nb - 1 ? large : nb - 1;
  }
  bucket += val;
  for (int hh = 0; hh < HN; ++hh)
    pb[((size_t)hh * T + i) * T + j] = re[(size_t)bucket * HN + hh];
}

// =====================================================================
// 10) gated relative-position gate: per (b,h,t)
// =====================================================================
__global__ void k_gate(const float* __restrict__ h, const float* __restrict__ gw,
                       const float* __restrict__ gb, const float* __restrict__ gc,
                       float* __restrict__ gate, int T) {
  int idx = blockIdx.x * blockDim.x + threadIdx.x;
  int total = BN * HN * T;
  if (idx >= total) return;
  int t = idx % T;
  int bh = idx / T;
  int hh = bh % HN;
  int b = bh / HN;
  const float* hv = h + ((size_t)b * T + t) * DM + hh * HD;
  float rp[8];
#pragma unroll
  for (int o = 0; o < 8; ++o) rp[o] = gb[o];
  for (int d = 0; d < HD; ++d) {
    float xv = hv[d];
#pragma unroll
    for (int o = 0; o < 8; ++o) rp[o] += xv * gw[d * 8 + o];
  }
  float s0 = rp[0] + rp[1] + rp[2] + rp[3];
  float s1 = rp[4] + rp[5] + rp[6] + rp[7];
  float g0 = sigmoid_f(s0), g1 = sigmoid_f(s1);
  gate[idx] = g0 * (g1 * gc[hh] - 1.0f) + 2.0f;
}

// =====================================================================
// 11) scores += gate * pos_bias, then row softmax (in place)
// =====================================================================
__global__ void k_attn_softmax(float* __restrict__ sc, const float* __restrict__ gate,
                               const float* __restrict__ pb, int T) {
  int row = blockIdx.x, tid = threadIdx.x;
  int bh = row / T;
  int i = row - bh * T;
  int hh = bh % HN;
  float g = gate[(size_t)bh * T + i];
  float* sr = sc + (size_t)row * T;
  const float* pr = pb + ((size_t)hh * T + i) * T;
  __shared__ float red[256];
  float mx = -3.0e38f;
  for (int j = tid; j < T; j += 256) { float v = sr[j] + g * pr[j]; sr[j] = v; mx = fmaxf(mx, v); }
  red[tid] = mx; __syncthreads();
  for (int st = 128; st > 0; st >>= 1) { if (tid < st) red[tid] = fmaxf(red[tid], red[tid + st]); __syncthreads(); }
  mx = red[0]; __syncthreads();
  float s = 0.f;
  for (int j = tid; j < T; j += 256) { float e = expf(sr[j] - mx); sr[j] = e; s += e; }
  red[tid] = s; __syncthreads();
  for (int st = 128; st > 0; st >>= 1) { if (tid < st) red[tid] += red[tid + st]; __syncthreads(); }
  float inv = 1.0f / red[0];
  for (int j = tid; j < T; j += 256) sr[j] *= inv;
}

// =====================================================================
// 12) small helpers: softmax over 13 layer weights, scale, axpy, pool
// =====================================================================
__global__ void k_softmax13(const float* __restrict__ x, float* __restrict__ y, int n) {
  if (threadIdx.x == 0 && blockIdx.x == 0) {
    float mx = -3.0e38f;
    for (int i = 0; i < n; ++i) mx = fmaxf(mx, x[i]);
    float s = 0.f;
    for (int i = 0; i < n; ++i) { float e = expf(x[i] - mx); y[i] = e; s += e; }
    for (int i = 0; i < n; ++i) y[i] /= s;
  }
}

__global__ void k_scale(float* __restrict__ y, const float* __restrict__ x,
                        const float* __restrict__ s, int si, long long n) {
  long long i = (long long)blockIdx.x * blockDim.x + threadIdx.x;
  if (i < n) y[i] = s[si] * x[i];
}

__global__ void k_axpy(float* __restrict__ y, const float* __restrict__ x,
                       const float* __restrict__ s, int si, long long n) {
  long long i = (long long)blockIdx.x * blockDim.x + threadIdx.x;
  if (i < n) y[i] += s[si] * x[i];
}

__global__ void k_pool(const float* __restrict__ acc, const int* __restrict__ mask,
                       float* __restrict__ pooled, int T) {
  int idx = blockIdx.x * blockDim.x + threadIdx.x;
  if (idx >= BN * DM) return;
  int b = idx / DM, d = idx % DM;
  float s = 0.f, ds = 0.f;
  for (int t = 0; t < T; ++t) {
    float m = (float)mask[(size_t)b * LW + (size_t)t * 320];
    ds += m;
    s += m * acc[((size_t)b * T + t) * DM + d];
  }
  pooled[idx] = s / fmaxf(ds, 1e-9f);
}

// =====================================================================
// host-side launch helper
// =====================================================================
static inline void launch_gemm(hipStream_t st, const float* A, int lda, long long sAi, long long sAo,
                               const float* B, int ldb, long long sBi, long long sBo, int transB,
                               const float* bias, float* C, int ldc, long long sCi, long long sCo,
                               int M, int N, int K, int batch, int inner, int act, float alpha) {
  dim3 g((M + 63) / 64, (N + 63) / 64, batch);
  k_gemm<<<g, 128, 0, st>>>(A, lda, sAi, sAo, B, ldb, sBi, sBo, transB, bias,
                            C, ldc, sCi, sCo, M, N, K, inner, act, alpha);
}

extern "C" void kernel_launch(void* const* d_in, const int* in_sizes, int n_in,
                              void* d_out, int out_size, void* d_ws, size_t ws_size,
                              hipStream_t stream) {
  (void)in_sizes; (void)n_in; (void)out_size; (void)ws_size;
  // ---- inputs in setup_inputs() dict order ----
  const float* wav  = (const float*)d_in[0];
  const int*   mask = (const int*)d_in[1];
  const float* convw[7];
  for (int i = 0; i < 7; ++i) convw[i] = (const float*)d_in[2 + i];
  const float* gn_w   = (const float*)d_in[9];
  const float* gn_b   = (const float*)d_in[10];
  const float* fpln_w = (const float*)d_in[11];
  const float* fpln_b = (const float*)d_in[12];
  const float* fp_w   = (const float*)d_in[13];
  const float* fp_b   = (const float*)d_in[14];
  const float* pcv_w  = (const float*)d_in[15];
  const float* pcv_b  = (const float*)d_in[16];
  const float* eln_w  = (const float*)d_in[17];
  const float* eln_b  = (const float*)d_in[18];
  const float* rel_e  = (const float*)d_in[19];
  const float* q_w  = (const float*)d_in[20];
  const float* q_b  = (const float*)d_in[21];
  const float* kk_w = (const float*)d_in[22];
  const float* kk_b = (const float*)d_in[23];
  const float* v_w  = (const float*)d_in[24];
  const float* v_b  = (const float*)d_in[25];
  const float* o_w  = (const float*)d_in[26];
  const float* o_b  = (const float*)d_in[27];
  const float* gru_w = (const float*)d_in[28];
  const float* gru_b = (const float*)d_in[29];
  const float* gru_c = (const float*)d_in[30];
  const float* ln1_w = (const float*)d_in[31];
  const float* ln1_b = (const float*)d_in[32];
  const float* ff1_w = (const float*)d_in[33];
  const float* ff1_b = (const float*)d_in[34];
  const float* ff2_w = (const float*)d_in[35];
  const float* ff2_b = (const float*)d_in[36];
  const float* ln2_w = (const float*)d_in[37];
  const float* ln2_b = (const float*)d_in[38];
  const float* lweights = (const float*)d_in[39];
  const float* c1_w = (const float*)d_in[40];
  const float* c1_b = (const float*)d_in[41];
  const float* cln_w = (const float*)d_in[42];
  const float* cln_b = (const float*)d_in[43];
  const float* c2_w = (const float*)d_in[44];
  const float* c2_b = (const float*)d_in[45];
  float* outp = (float*)d_out;

  // ---- workspace layout ----
  float* W = (float*)d_ws;
  size_t off = 0;
  auto alloc = [&](size_t n) { float* p = W + off; off += n; return p; };
  float* wavn = alloc((size_t)BN * LW);
  float* bufA = alloc((size_t)CD * BN * 15999);   // conv ping (also hosts temps later)
  float* bufB = alloc((size_t)CD * BN * 7999);    // conv pong
  // temps packed inside bufA after the final conv output region
  size_t o2 = (size_t)CD * BN * TT + 4096;
  auto alloc2 = [&](size_t n) { float* p = bufA + o2; o2 += n; return p; };
  float* xt     = alloc2((size_t)ROWS * CD);
  float* xln    = alloc2((size_t)ROWS * CD);
  float* hbuf   = alloc2((size_t)ROWS * DM);
  float* pcb    = alloc2((size_t)ROWS * DM);
  float* henc   = alloc2((size_t)ROWS * DM);
  float* pb     = alloc2((size_t)HN * TT * TT);
  float* lwb    = alloc2(16);
  float* accum  = alloc2((size_t)ROWS * DM);
  float* qbuf   = alloc2((size_t)ROWS * DM);
  float* kbuf   = alloc2((size_t)ROWS * DM);
  float* vbuf   = alloc2((size_t)ROWS * DM);
  float* scores = alloc2((size_t)BN * HN * TT * TT);
  float* atout  = alloc2((size_t)ROWS * DM);
  float* x1b    = alloc2((size_t)ROWS * DM);
  float* ffb    = alloc2((size_t)ROWS * FFN);
  float* x2b    = alloc2((size_t)ROWS * DM);
  float* gateb  = alloc2((size_t)BN * HN * TT);
  float* pooled = alloc2((size_t)BN * DM);
  float* cls1   = alloc2((size_t)BN * 512);
  float* cls2   = alloc2((size_t)BN * 512);

  // ---- 1. waveform norm ----
  k_wav_norm<<<BN, 256, 0, stream>>>(wav, wavn, LW);

  // ---- 2. conv0 + instance-norm + gelu ----
  int t1 = 15999;
  {
    dim3 g((t1 + 255) / 256, CD, BN);
    k_conv0<<<g, 256, 0, stream>>>(wavn, convw[0], bufA, LW, t1);
  }
  k_instnorm_gelu<<<BN * CD, 256, 0, stream>>>(bufA, gn_w, gn_b, t1);

  // ---- 3. conv layers 2..7 via WMMA (im2col) + gelu ----
  {
    const int kz[7] = {10, 3, 3, 3, 3, 2, 2};
    int tin = t1;
    float* src = bufA;
    float* dst = bufB;
    for (int i = 1; i < 7; ++i) {
      int kc = kz[i];
      int tout = (tin - kc) / 2 + 1;
      dim3 g((tout + 31) / 32, CD / 32, BN);
      k_conv_wmma_gelu<<<g, 32, 0, stream>>>(src, convw[i], dst, tin, tout, kc, 2);
      float* tmp = src; src = dst; dst = tmp;
      tin = tout;
    }
    // final result lands in bufA, tin == TT
  }

  // ---- 4. feature projection ----
  {
    dim3 g(TT, BN);
    k_transpose_ct<<<g, CD, 0, stream>>>(bufA, xt, TT);
  }
  k_layernorm<<<ROWS, 256, 0, stream>>>(xt, nullptr, fpln_w, fpln_b, xln, CD, 0);
  launch_gemm(stream, xln, CD, 0, 0, fp_w, DM, 0, 0, 0, fp_b,
              hbuf, DM, 0, 0, ROWS, DM, CD, 1, 1, 0, 1.0f);

  // ---- 5. positional conv + encoder layernorm ----
  {
    dim3 g(TT, BN);
    k_posconv_gelu<<<g, DM, 0, stream>>>(hbuf, pcv_w, pcv_b, pcb, TT);
  }
  k_layernorm<<<ROWS, 256, 0, stream>>>(hbuf, pcb, eln_w, eln_b, henc, DM, 0);

  // ---- 6. relative position bias + layer-weight softmax ----
  {
    dim3 g((TT + 255) / 256, TT);
    k_posbias<<<g, 256, 0, stream>>>(rel_e, pb, TT);
  }
  k_softmax13<<<1, 32, 0, stream>>>(lweights, lwb, NL + 1);

  const long long nElem = (long long)ROWS * DM;
  int eBlocks = (int)((nElem + 255) / 256);
  k_scale<<<eBlocks, 256, 0, stream>>>(accum, henc, lwb, 0, nElem);

  // ---- 7. transformer layers ----
  const float scaleq = 0.125f;  // 1/sqrt(64)
  float* hcur = henc;
  for (int l = 0; l < NL; ++l) {
    const float* qw = q_w  + (size_t)l * DM * DM;  const float* qb = q_b  + (size_t)l * DM;
    const float* kw = kk_w + (size_t)l * DM * DM;  const float* kb = kk_b + (size_t)l * DM;
    const float* vw = v_w  + (size_t)l * DM * DM;  const float* vb = v_b  + (size_t)l * DM;
    const float* ow = o_w  + (size_t)l * DM * DM;  const float* ob = o_b  + (size_t)l * DM;
    const float* gw = gru_w + (size_t)l * HD * 8;  const float* gb = gru_b + (size_t)l * 8;
    const float* gc = gru_c + (size_t)l * HN;
    const float* l1w = ln1_w + (size_t)l * DM;     const float* l1b = ln1_b + (size_t)l * DM;
    const float* f1w = ff1_w + (size_t)l * DM * FFN; const float* f1b = ff1_b + (size_t)l * FFN;
    const float* f2w = ff2_w + (size_t)l * FFN * DM; const float* f2b = ff2_b + (size_t)l * DM;
    const float* l2w = ln2_w + (size_t)l * DM;     const float* l2b = ln2_b + (size_t)l * DM;

    // q (pre-scaled), k, v
    launch_gemm(stream, hcur, DM, 0, 0, qw, DM, 0, 0, 0, qb, qbuf, DM, 0, 0,
                ROWS, DM, DM, 1, 1, 0, scaleq);
    launch_gemm(stream, hcur, DM, 0, 0, kw, DM, 0, 0, 0, kb, kbuf, DM, 0, 0,
                ROWS, DM, DM, 1, 1, 0, 1.0f);
    launch_gemm(stream, hcur, DM, 0, 0, vw, DM, 0, 0, 0, vb, vbuf, DM, 0, 0,
                ROWS, DM, DM, 1, 1, 0, 1.0f);
    // gate
    {
      int total = BN * HN * TT;
      k_gate<<<(total + 255) / 256, 256, 0, stream>>>(hcur, gw, gb, gc, gateb, TT);
    }
    // scores = q . k^T   (batched over 96 (b,h))
    launch_gemm(stream,
                qbuf, DM, 64, (long long)TT * DM,
                kbuf, DM, 64, (long long)TT * DM, 1, nullptr,
                scores, TT, (long long)TT * TT, (long long)HN * TT * TT,
                TT, TT, HD, BN * HN, HN, 0, 1.0f);
    // + gate * pos_bias ; softmax rows
    k_attn_softmax<<<BN * HN * TT, 256, 0, stream>>>(scores, gateb, pb, TT);
    // context = attn . v
    launch_gemm(stream,
                scores, TT, (long long)TT * TT, (long long)HN * TT * TT,
                vbuf, DM, 64, (long long)TT * DM, 0, nullptr,
                atout, DM, 64, (long long)TT * DM,
                TT, HD, TT, BN * HN, HN, 0, 1.0f);
    // output projection (reuse qbuf)
    launch_gemm(stream, atout, DM, 0, 0, ow, DM, 0, 0, 0, ob, qbuf, DM, 0, 0,
                ROWS, DM, DM, 1, 1, 0, 1.0f);
    // x1 = LN(h + attn_out)
    k_layernorm<<<ROWS, 256, 0, stream>>>(hcur, qbuf, l1w, l1b, x1b, DM, 0);
    // ff = gelu(x1 @ ff1 + b1)
    launch_gemm(stream, x1b, DM, 0, 0, f1w, FFN, 0, 0, 0, f1b, ffb, FFN, 0, 0,
                ROWS, FFN, DM, 1, 1, 1, 1.0f);
    // ff2 (reuse kbuf)
    launch_gemm(stream, ffb, FFN, 0, 0, f2w, DM, 0, 0, 0, f2b, kbuf, DM, 0, 0,
                ROWS, DM, FFN, 1, 1, 0, 1.0f);
    // x2 = LN(x1 + ff2)
    k_layernorm<<<ROWS, 256, 0, stream>>>(x1b, kbuf, l2w, l2b, x2b, DM, 0);
    // accum += lw[l+1] * x2
    k_axpy<<<eBlocks, 256, 0, stream>>>(accum, x2b, lwb, l + 1, nElem);
    hcur = x2b;
  }

  // ---- 8. masked mean pooling ----
  k_pool<<<(BN * DM + 255) / 256, 256, 0, stream>>>(accum, mask, pooled, TT);

  // ---- 9. classifier head ----
  launch_gemm(stream, pooled, DM, 0, 0, c1_w, 512, 0, 0, 0, c1_b, cls1, 512, 0, 0,
              BN, 512, DM, 1, 1, 0, 1.0f);
  k_layernorm<<<BN, 256, 0, stream>>>(cls1, nullptr, cln_w, cln_b, cls2, 512, 2); // relu(LN)
  launch_gemm(stream, cls2, 512, 0, 0, c2_w, 7, 0, 0, 0, c2_b, outp, 7, 0, 0,
              BN, 7, 512, 1, 1, 0, 1.0f);
}
